// LSTMPolicy_81939386073147
// MI455X (gfx1250) — matrix operand order repaired and checked by reading, not compiled, not run on hardware
//
#include <hip/hip_runtime.h>
#include <hip/hip_bf16.h>

// ---------------------------------------------------------------------------
// LSTM policy network for MI455X (gfx1250), wave32 + WMMA f32_16x16x32_f16.
// B=512, S=512, OBS=64, H=128, A=9, L=2
// LSTM layer kernel keeps Wih+Whh (256 KB f16) resident in CDNA5's 320 KB LDS,
// pre-swizzled into WMMA B-fragment order.
// ---------------------------------------------------------------------------

typedef __attribute__((ext_vector_type(16))) _Float16 v16h;
typedef __attribute__((ext_vector_type(8)))  _Float16 v8h;
typedef __attribute__((ext_vector_type(8)))  float    v8f;

#define DEVINL __device__ __forceinline__

static constexpr int   Bn   = 512;
static constexpr int   Sn   = 512;
static constexpr int   Hn   = 128;
static constexpr int   An   = 9;
static constexpr long  BS   = (long)Bn * Sn;       // 262144 tokens

// ---- WMMA helpers ----------------------------------------------------------

DEVINL v8f wmma_f16(v16h a, v16h b, v8f c) {
  // 8 args: (neg_a, A, neg_b, B, c_mod, C, reuse_a, reuse_b)
  return __builtin_amdgcn_wmma_f32_16x16x32_f16(false, a, false, b, (short)0, c,
                                                false, false);
}

// A-matrix 16x32 f16 fragment: lane holds row m = lane&15,
// half hf = lane>>4; elements 0..15 = K[k0+hf*16 .. +15] (contiguous, matches B).
// `prow` = per-lane row pointer.
DEVINL v16h load_a(const _Float16* prow, int k0, int hf) {
  return *(const v16h*)(prow + k0 + hf * 16);
}

// B-matrix 32x16 f16 fragment from row-major W (Ntot x Ktot):
// lane holds column n = lane&15, K = (lane>>4)*16 + e  (16 contiguous halves).
DEVINL v16h load_b(const _Float16* __restrict__ W, int ldk, int n, int k0, int hf) {
  return *(const v16h*)(W + (long)n * ldk + k0 + hf * 16);
}

// B-fragment from LDS, pre-swizzled: frag[ntile][kc][lane][0..15] contiguous.
DEVINL v16h load_b_sw(const _Float16* sW, int ntg, int kc, int lane) {
  return *(const v16h*)(sW + (((ntg * 4 + kc) * 32 + lane) << 4));
}

DEVINL float sigmoidf(float x) { return 1.0f / (1.0f + __expf(-x)); }
DEVINL float fast_tanh(float x) {           // one v_exp + one division
  float e = __expf(2.0f * x);
  return (e - 1.0f) / (e + 1.0f);
}

// ---- workspace layout (bytes) ---------------------------------------------
static constexpr size_t OFF_ENC = 0;                             // enc  f16 64MiB
static constexpr size_t OFF_H0  = (size_t)BS * Hn * 2;           // h0   f16 64MiB
static constexpr size_t OFF_O1  = OFF_H0 * 2;                    // out1 f16 64MiB
static constexpr size_t OFF_W16 = OFF_H0 * 3;                    // packed f16 weights
static constexpr size_t OFF_F32 = OFF_W16 + 303104ull * 2;       // fused f32 biases

// f16-element offsets inside w16
static constexpr int W1E  = 0;        // enc_w1  (128 x 64)
static constexpr int W2E  = 8192;     // enc_w2  (128 x 128)
static constexpr int WIH0 = 24576;    // wih0    (512 x 128)
static constexpr int WHH0 = 90112;
static constexpr int WIH1 = 155648;
static constexpr int WHH1 = 221184;
static constexpr int HW1  = 286720;   // [am_w1 ; cr_w1] (128 x 128)
// float offsets inside f32x
static constexpr int BIAS0 = 0;       // bih0+bhh0 (512)
static constexpr int BIAS1 = 512;     // bih1+bhh1 (512)
static constexpr int HB1   = 1024;    // [am_b1 ; cr_b1] (128)

// dynamic LDS layout for the LSTM kernel (halves / floats):
//   sWih 65536 h | sWhh 65536 h | sHa 2048 h | sH 2048 f | sC 2048 f
//   sG 8192 f | sMask 16 f   => 315,456 bytes  (<= 320 KB CDNA5 LDS)
static constexpr size_t LSTM_LDS_BYTES =
    65536 * 2 * 2 + 2048 * 2 + 2048 * 4 + 2048 * 4 + 8192 * 4 + 16 * 4;

// ---------------------------------------------------------------------------
// Kernel 1: weight repack f32 -> f16 + bias fusion.
// ---------------------------------------------------------------------------
__global__ void prep_kernel(const float* __restrict__ enc_w1,
                            const float* __restrict__ enc_w2,
                            const float* __restrict__ wih0,
                            const float* __restrict__ whh0,
                            const float* __restrict__ wih1,
                            const float* __restrict__ whh1,
                            const float* __restrict__ am_w1,
                            const float* __restrict__ cr_w1,
                            const float* __restrict__ bih0,
                            const float* __restrict__ bhh0,
                            const float* __restrict__ bih1,
                            const float* __restrict__ bhh1,
                            const float* __restrict__ am_b1,
                            const float* __restrict__ cr_b1,
                            _Float16* __restrict__ w16,
                            float* __restrict__ f32x) {
  const int total = 303104 + 1152;
  for (int i = blockIdx.x * blockDim.x + threadIdx.x; i < total;
       i += gridDim.x * blockDim.x) {
    if (i < 303104) {
      float v;
      if      (i < W2E)   v = enc_w1[i];
      else if (i < WIH0)  v = enc_w2[i - W2E];
      else if (i < WHH0)  v = wih0[i - WIH0];
      else if (i < WIH1)  v = whh0[i - WHH0];
      else if (i < WHH1)  v = wih1[i - WIH1];
      else if (i < HW1)   v = whh1[i - WHH1];
      else {
        int k = i - HW1;
        v = (k < 8192) ? am_w1[k] : cr_w1[k - 8192];
      }
      w16[i] = (_Float16)v;
    } else {
      int j = i - 303104;
      float v;
      if      (j < 512)  v = bih0[j] + bhh0[j];
      else if (j < 1024) v = bih1[j - 512] + bhh1[j - 512];
      else {
        int k = j - 1024;
        v = (k < 64) ? am_b1[k] : cr_b1[k - 64];
      }
      f32x[j] = v;
    }
  }
}

// ---------------------------------------------------------------------------
// Kernel 2: encoder.  One WG = 64 tokens, 256 threads = 8 waves.
//   GEMM1 (64x64 @ 64x128) -> LayerNorm -> ReLU -> GEMM2 (64x128 @ 128x128)
// ---------------------------------------------------------------------------
__global__ __launch_bounds__(256) void enc_kernel(
    const float* __restrict__ obs, const float* __restrict__ b1,
    const float* __restrict__ lng, const float* __restrict__ lnb,
    const float* __restrict__ b2, const _Float16* __restrict__ w1,
    const _Float16* __restrict__ w2, _Float16* __restrict__ enc_out) {
  __shared__ _Float16 sObs[64 * 64];
  __shared__ float    sY[64 * 132];     // padded: avoids LN bank conflicts
  __shared__ _Float16 sA[64 * 128];
  __shared__ float    sMu[64], sRs[64];

  const int  tid   = threadIdx.x;
  const int  lane  = tid & 31, wv = tid >> 5;
  const int  hf    = lane >> 4, ln15 = lane & 15;
  const long tok0  = (long)blockIdx.x * 64;

  // stage obs tile as f16
#pragma unroll
  for (int it = 0; it < 16; ++it) {
    int idx = tid + it * 256;
    sObs[idx] = (_Float16)obs[tok0 * 64 + idx];
  }
  __syncthreads();

  const int rt = wv >> 1;            // row-tile 0..3 (16 tokens each)
  const int nb = (wv & 1) * 64;      // half of the 128 output cols

  // ---- GEMM1 + bias -> sY
  {
    const _Float16* arow = &sObs[(rt * 16 + ln15) * 64];
    v16h a0 = load_a(arow, 0, hf);
    v16h a1 = load_a(arow, 32, hf);
#pragma unroll
    for (int nt = 0; nt < 4; ++nt) {
      int   n  = nb + nt * 16;
      float bv = b1[n + ln15];
      v8f acc;
#pragma unroll
      for (int i = 0; i < 8; ++i) acc[i] = bv;
      acc = wmma_f16(a0, load_b(w1, 64, n + ln15, 0, hf), acc);
      acc = wmma_f16(a1, load_b(w1, 64, n + ln15, 32, hf), acc);
#pragma unroll
      for (int j = 0; j < 8; ++j)
        sY[(rt * 16 + j + 8 * hf) * 132 + n + ln15] = acc[j];
    }
  }
  __syncthreads();

  // ---- LayerNorm stats (one thread per token)
  if (tid < 64) {
    const float* row = &sY[tid * 132];
    float s = 0.f, s2 = 0.f;
    for (int j = 0; j < 128; ++j) { float v = row[j]; s += v; s2 += v * v; }
    float mu  = s * (1.0f / 128.0f);
    float var = s2 * (1.0f / 128.0f) - mu * mu;
    sMu[tid] = mu;
    sRs[tid] = rsqrtf(var + 1e-5f);
  }
  __syncthreads();

  // ---- normalize + affine + ReLU -> f16
#pragma unroll
  for (int it = 0; it < 32; ++it) {
    int   idx = tid + it * 256;
    int   tk  = idx >> 7, j = idx & 127;
    float v   = (sY[tk * 132 + j] - sMu[tk]) * sRs[tk] * lng[j] + lnb[j];
    sA[idx]   = (_Float16)fmaxf(v, 0.0f);
  }
  __syncthreads();

  // ---- GEMM2 + bias + ReLU -> enc_out (f16)
  {
    const _Float16* arow = &sA[(rt * 16 + ln15) * 128];
    v16h a[4];
#pragma unroll
    for (int kc = 0; kc < 4; ++kc) a[kc] = load_a(arow, kc * 32, hf);
#pragma unroll
    for (int nt = 0; nt < 4; ++nt) {
      int   n  = nb + nt * 16;
      float bv = b2[n + ln15];
      v8f acc;
#pragma unroll
      for (int i = 0; i < 8; ++i) acc[i] = bv;
#pragma unroll
      for (int kc = 0; kc < 4; ++kc)
        acc = wmma_f16(a[kc], load_b(w2, 128, n + ln15, kc * 32, hf), acc);
#pragma unroll
      for (int j = 0; j < 8; ++j) {
        long row = tok0 + rt * 16 + j + 8 * hf;
        enc_out[row * 128 + n + ln15] = (_Float16)fmaxf(acc[j], 0.0f);
      }
    }
  }
}

// ---------------------------------------------------------------------------
// Kernel 3: one LSTM layer, full time loop.  Batch rows are independent:
// WG = 16 batch rows, 512 threads = 16 waves (each wave owns 32 gate cols).
// Wih/Whh stay resident in LDS (256 KB), pre-swizzled into per-lane WMMA
// B-fragment order => steady-state global traffic is just the x row + mask.
// ---------------------------------------------------------------------------
__global__ __launch_bounds__(512) void lstm_kernel(
    const _Float16* __restrict__ xin,   // (B,S,H) f16
    const _Float16* __restrict__ wih,   // (512,128) f16 row-major
    const _Float16* __restrict__ whh,   // (512,128) f16 row-major
    const float* __restrict__ bias,     // (512) = bih+bhh
    const float* __restrict__ masks,    // (B,S)
    _Float16* __restrict__ hout,        // (B,S,H) f16
    float* __restrict__ hfin,           // (B,H) f32
    float* __restrict__ cfin) {         // (B,H) f32
  extern __shared__ char smem[];
  _Float16* sWih  = (_Float16*)smem;             // 65536 halves (swizzled)
  _Float16* sWhh  = sWih + 65536;                // 65536 halves (swizzled)
  _Float16* sHa   = sWhh + 65536;                // 16x128 f16 A operand
  float*    sH    = (float*)(sHa + 2048);        // 16x128
  float*    sC    = sH + 2048;                   // 16x128
  float*    sG    = sC + 2048;                   // 16x512 gate pre-acts
  float*    sMask = sG + 8192;                   // 16

  const int tid  = threadIdx.x;
  const int lane = tid & 31, wv = tid >> 5;
  const int hf   = lane >> 4, ln15 = lane & 15;
  const int r0   = blockIdx.x * 16;

  // stage weights into LDS in fragment order:
  //   sW[((nt*4+kc)*32+lane)*16 + e] = W[nt*16+(lane&15)][kc*32+(lane>>4)*16+e]
  for (int i = tid; i < 65536; i += 512) {
    int e = i & 15, ln = (i >> 4) & 31, kc = (i >> 9) & 3, nt = i >> 11;
    int n = nt * 16 + (ln & 15);
    int k = kc * 32 + (ln >> 4) * 16 + e;
    sWih[i] = wih[n * 128 + k];
    sWhh[i] = whh[n * 128 + k];
  }
  for (int idx = tid; idx < 2048; idx += 512) { sH[idx] = 0.f; sC[idx] = 0.f; }
  if (tid < 16) sMask[tid] = masks[(long)(r0 + tid) * Sn];
  __syncthreads();

  const _Float16* xrow = xin + (long)(r0 + ln15) * Sn * Hn;  // per-lane row ptr
  const float bv0 = bias[wv * 32 + ln15];
  const float bv1 = bias[wv * 32 + 16 + ln15];

  for (int t = 0; t < Sn; ++t) {
    // phase 0: mask reset, convert h -> f16 A buffer
    for (int idx = tid; idx < 2048; idx += 512) {
      int   m  = idx >> 7;
      float mk = sMask[m];
      float h  = sH[idx] * mk;
      sH[idx]  = h;
      sC[idx] *= mk;
      sHa[idx] = (_Float16)h;
    }
    __syncthreads();

    // phase 1: WMMA gate GEMM (x@Wih^T + h@Whh^T + bias)
    {
      const _Float16* xp = xrow + (long)t * Hn;
      __builtin_prefetch(xp + Hn, 0, 1);   // next timestep's x row
      v16h ax[4], ah[4];
#pragma unroll
      for (int kc = 0; kc < 4; ++kc) {
        ax[kc] = load_a(xp, kc * 32, hf);
        ah[kc] = load_a(&sHa[ln15 * 128], kc * 32, hf);
      }
#pragma unroll
      for (int nt = 0; nt < 2; ++nt) {
        int   ntg = wv * 2 + nt;          // global n-tile 0..31
        float bv  = nt ? bv1 : bv0;
        v8f acc;
#pragma unroll
        for (int i = 0; i < 8; ++i) acc[i] = bv;
#pragma unroll
        for (int kc = 0; kc < 4; ++kc) {
          acc = wmma_f16(ax[kc], load_b_sw(sWih, ntg, kc, lane), acc);
          acc = wmma_f16(ah[kc], load_b_sw(sWhh, ntg, kc, lane), acc);
        }
#pragma unroll
        for (int j = 0; j < 8; ++j)
          sG[(j + 8 * hf) * 512 + ntg * 16 + ln15] = acc[j];
      }
    }
    __syncthreads();

    // phase 2: activations, state update, h stream-out; preload next mask
    for (int idx = tid; idx < 2048; idx += 512) {
      int          m  = idx >> 7, j = idx & 127;
      const float* gm = &sG[m * 512];
      float gi = gm[j], gf = gm[128 + j], gg = gm[256 + j], go = gm[384 + j];
      float c  = sigmoidf(gf) * sC[idx] + sigmoidf(gi) * fast_tanh(gg);
      float h  = sigmoidf(go) * fast_tanh(c);
      sC[idx]  = c;
      sH[idx]  = h;
      hout[((long)(r0 + m) * Sn + t) * Hn + j] = (_Float16)h;
    }
    if (tid < 16 && t + 1 < Sn)
      sMask[tid] = masks[(long)(r0 + tid) * Sn + t + 1];
    __syncthreads();
  }

  // final carry out (post-step at t = S-1, no mask)
  for (int idx = tid; idx < 2048; idx += 512) {
    int m = idx >> 7, j = idx & 127;
    hfin[(long)(r0 + m) * Hn + j] = sH[idx];
    cfin[(long)(r0 + m) * Hn + j] = sC[idx];
  }
}

// ---------------------------------------------------------------------------
// Kernel 4: actor/critic heads.  WG = 64 tokens, 256 threads = 8 waves.
// ---------------------------------------------------------------------------
__global__ __launch_bounds__(256) void heads_kernel(
    const _Float16* __restrict__ lo,     // lstm_out (B,S,H) f16
    const _Float16* __restrict__ hw1,    // (128,128) f16
    const float* __restrict__ hb1,       // (128)
    const float* __restrict__ am_w2,     // (9,64)
    const float* __restrict__ am_b2,     // (9)
    const float* __restrict__ log_std,   // (9)
    const float* __restrict__ cr_w2,     // (64)
    const float* __restrict__ cr_b2,     // (1)
    const float* __restrict__ noise,     // (B,S,9)
    float* __restrict__ action, float* __restrict__ logp,
    float* __restrict__ value) {
  __shared__ float sHid[64 * 130];       // padded stride vs LDS bank conflicts

  const int  tid  = threadIdx.x;
  const int  lane = tid & 31, wv = tid >> 5;
  const int  hf   = lane >> 4, ln15 = lane & 15;
  const long tok0 = (long)blockIdx.x * 64;
  const int  rt   = wv >> 1, nb = (wv & 1) * 64;

  {
    const _Float16* arow = lo + (tok0 + rt * 16 + ln15) * Hn;
    v16h a[4];
#pragma unroll
    for (int kc = 0; kc < 4; ++kc) a[kc] = load_a(arow, kc * 32, hf);
#pragma unroll
    for (int nt = 0; nt < 4; ++nt) {
      int   n  = nb + nt * 16;
      float bv = hb1[n + ln15];
      v8f acc;
#pragma unroll
      for (int i = 0; i < 8; ++i) acc[i] = bv;
#pragma unroll
      for (int kc = 0; kc < 4; ++kc)
        acc = wmma_f16(a[kc], load_b(hw1, 128, n + ln15, kc * 32, hf), acc);
#pragma unroll
      for (int j = 0; j < 8; ++j)
        sHid[(rt * 16 + j + 8 * hf) * 130 + n + ln15] = fmaxf(acc[j], 0.0f);
    }
  }
  __syncthreads();

  if (tid < 64) {
    const float* hid = &sHid[tid * 130];
    const long   tok = tok0 + tid;
    float lp = 0.0f;
#pragma unroll
    for (int aI = 0; aI < An; ++aI) {
      float s = am_b2[aI];
      for (int k = 0; k < 64; ++k) s += hid[k] * am_w2[aI * 64 + k];
      float mean = fast_tanh(s);
      float ls   = fminf(fmaxf(log_std[aI], -4.0f), 0.0f);
      float std  = __expf(ls);
      float nz   = noise[tok * An + aI];
      action[tok * An + aI] = mean + std * nz;
      lp += -0.5f * nz * nz - ls;
    }
    lp -= (float)An * 0.91893853320467274178f;  // A * 0.5*ln(2*pi)
    logp[tok] = lp;
    float v = cr_b2[0];
    for (int k = 0; k < 64; ++k) v += hid[64 + k] * cr_w2[k];
    value[tok] = v;
  }
}

// ---------------------------------------------------------------------------
extern "C" void kernel_launch(void* const* d_in, const int* in_sizes, int n_in,
                              void* d_out, int out_size, void* d_ws,
                              size_t ws_size, hipStream_t stream) {
  (void)in_sizes; (void)n_in; (void)out_size; (void)ws_size;

  const float* obs     = (const float*)d_in[0];
  const float* masks   = (const float*)d_in[1];
  const float* noise   = (const float*)d_in[2];
  const float* enc_w1  = (const float*)d_in[3];
  const float* enc_b1  = (const float*)d_in[4];
  const float* ln_g    = (const float*)d_in[5];
  const float* ln_b    = (const float*)d_in[6];
  const float* enc_w2  = (const float*)d_in[7];
  const float* enc_b2  = (const float*)d_in[8];
  const float* wih0    = (const float*)d_in[9];
  const float* whh0    = (const float*)d_in[10];
  const float* bih0    = (const float*)d_in[11];
  const float* bhh0    = (const float*)d_in[12];
  const float* wih1    = (const float*)d_in[13];
  const float* whh1    = (const float*)d_in[14];
  const float* bih1    = (const float*)d_in[15];
  const float* bhh1    = (const float*)d_in[16];
  const float* am_w1   = (const float*)d_in[17];
  const float* am_b1   = (const float*)d_in[18];
  const float* am_w2   = (const float*)d_in[19];
  const float* am_b2   = (const float*)d_in[20];
  const float* log_std = (const float*)d_in[21];
  const float* cr_w1   = (const float*)d_in[22];
  const float* cr_b1   = (const float*)d_in[23];
  const float* cr_w2   = (const float*)d_in[24];
  const float* cr_b2   = (const float*)d_in[25];

  char* ws = (char*)d_ws;
  _Float16* encB = (_Float16*)(ws + OFF_ENC);
  _Float16* h0B  = (_Float16*)(ws + OFF_H0);
  _Float16* o1B  = (_Float16*)(ws + OFF_O1);
  _Float16* w16  = (_Float16*)(ws + OFF_W16);
  float*    f32x = (float*)(ws + OFF_F32);

  float* out    = (float*)d_out;
  float* action = out;                         // (B,S,A)  2,359,296
  float* lpO    = action + (long)BS * An;      // (B,S)      262,144
  float* valO   = lpO + BS;                    // (B,S)      262,144
  float* hO     = valO + BS;                   // (L,B,H)    131,072
  float* cO     = hO + 2L * Bn * Hn;           // (L,B,H)    131,072

  // allow the 308 KB dynamic-LDS request (CDNA5 WGs may use up to 320 KB)
  (void)hipFuncSetAttribute((const void*)lstm_kernel,
                            hipFuncAttributeMaxDynamicSharedMemorySize,
                            (int)LSTM_LDS_BYTES);

  prep_kernel<<<dim3(296), dim3(256), 0, stream>>>(
      enc_w1, enc_w2, wih0, whh0, wih1, whh1, am_w1, cr_w1, bih0, bhh0, bih1,
      bhh1, am_b1, cr_b1, w16, f32x);

  enc_kernel<<<dim3(BS / 64), dim3(256), 0, stream>>>(
      obs, enc_b1, ln_g, ln_b, enc_b2, w16 + W1E, w16 + W2E, encB);

  lstm_kernel<<<dim3(Bn / 16), dim3(512), LSTM_LDS_BYTES, stream>>>(
      encB, w16 + WIH0, w16 + WHH0, f32x + BIAS0, masks, h0B, hO, cO);

  lstm_kernel<<<dim3(Bn / 16), dim3(512), LSTM_LDS_BYTES, stream>>>(
      h0B, w16 + WIH1, w16 + WHH1, f32x + BIAS1, masks, o1B,
      hO + (long)Bn * Hn, cO + (long)Bn * Hn);

  heads_kernel<<<dim3(BS / 64), dim3(256), 0, stream>>>(
      o1B, w16 + HW1, f32x + HB1, am_w2, am_b2, log_std, cr_w2, cr_b2, noise,
      action, lpO, valO);
}
